// GraphNetSoftMax_86535001079872
// MI455X (gfx1250) — matrix operand order repaired and checked by reading, not compiled, not run on hardware
//
#include <hip/hip_runtime.h>

// ---------------------------------------------------------------------------
// GraphNetSoftMax on MI455X (gfx1250, wave32, WMMA).
//
// N=100000 nodes, E=640000 edges, D=64, G=64 graphs, OUT=10.
// Heavy GEMMs ([rows,64]x[64,64]) run as v_wmma_f32_16x16x32_f16 tiles
// (f16 in, f32 accumulate).  BatchNorm batch stats are per-column sum/sumsq
// side-outputs of each GEMM: LDS ds_add_f32 block reduction, then one
// global_atomic_add_f32 flush per block.  The segment-sum uses a device-built
// CSR (by dst) so mailbox aggregation is an atomic-free coalesced gather.
// ---------------------------------------------------------------------------

namespace {
constexpr int   kN   = 100000;
constexpr int   kE   = 640000;
constexpr int   kG   = 64;
constexpr int   kOut = 10;
constexpr float kEps = 1e-5f;
}

typedef __attribute__((ext_vector_type(16))) _Float16 v16h;
typedef __attribute__((ext_vector_type(8)))  _Float16 v8h;
typedef __attribute__((ext_vector_type(8)))  float    v8f;
typedef __attribute__((ext_vector_type(4)))  float    v4f;

__device__ __forceinline__ float lrelu(float x) { return x > 0.f ? x : 0.2f * x; }

// -------------------------- small utility kernels --------------------------

__global__ void k_zero(float* p, int n) {
  int i = blockIdx.x * blockDim.x + threadIdx.x;
  if (i < n) p[i] = 0.f;
}

// W row-major [64(din),64(dout)] f32 -> Wt [n][k] f16 (B-frag friendly)
__global__ void k_w_to_f16t(const float* __restrict__ W, _Float16* __restrict__ Wt) {
  int i = blockIdx.x * blockDim.x + threadIdx.x;
  if (i < 64 * 64) {
    int n = i >> 6, k = i & 63;
    Wt[n * 64 + k] = (_Float16)W[k * 64 + n];
  }
}

// mean/var -> affine (scale, shift) for y' = y*scale + shift
__global__ void k_bn_finalize(const float* __restrict__ cs, const float* __restrict__ cq,
                              const float* __restrict__ gamma, const float* __restrict__ beta,
                              float* __restrict__ scale, float* __restrict__ shift, float cnt) {
  int j = threadIdx.x;
  if (j < 64) {
    float mean = cs[j] / cnt;
    float var  = cq[j] / cnt - mean * mean;
    float inv  = rsqrtf(var + kEps);
    float sc   = gamma[j] * inv;
    scale[j] = sc;
    shift[j] = beta[j] - mean * sc;
  }
}

// ------------------------- CSR (by dst) construction ------------------------

__global__ void k_hist(const int* __restrict__ dst, int* __restrict__ cnt) {
  int e = blockIdx.x * blockDim.x + threadIdx.x;
  if (e < kE) atomicAdd(&cnt[dst[e]], 1);
}

// single-block exclusive scan over kN counts -> off[kN+1], cur = copy of off
__global__ __launch_bounds__(1024) void k_scan(const int* __restrict__ cnt,
                                               int* __restrict__ off, int* __restrict__ cur) {
  __shared__ int part[1024];
  int t = threadIdx.x;
  constexpr int CH = (kN + 1023) / 1024;  // 98 items per thread
  int base = t * CH;
  int s = 0;
  for (int i = 0; i < CH; ++i) { int idx = base + i; if (idx < kN) s += cnt[idx]; }
  part[t] = s;
  __syncthreads();
  for (int d = 1; d < 1024; d <<= 1) {          // Hillis-Steele inclusive scan
    int v = (t >= d) ? part[t - d] : 0;
    __syncthreads();
    part[t] += v;
    __syncthreads();
  }
  int run = (t == 0) ? 0 : part[t - 1];
  for (int i = 0; i < CH; ++i) {
    int idx = base + i;
    if (idx < kN) { off[idx] = run; cur[idx] = run; run += cnt[idx]; }
  }
  if (t == 1023) off[kN] = part[1023];
}

__global__ void k_fill(const int* __restrict__ dst, int* __restrict__ cur,
                       int* __restrict__ el) {
  int e = blockIdx.x * blockDim.x + threadIdx.x;
  if (e < kE) {
    int p = atomicAdd(&cur[dst[e]], 1);
    el[p] = e;
  }
}

// ----------------------- WMMA fragment constructors ------------------------
// 16-bit A-matrix 16x32 layout (ISA 7.12.2):
//  lanes 0-15 : V0..3 = K0..7,  V4..7 = K16..23   (row M = lane%16)
//  lanes 16-31: V0..3 = K8..15, V4..7 = K24..31
// -> per frag, lane loads 8 contiguous f32 at (kbase+koff) and (kbase+16+koff),
//    koff = (lane>=16)*8.  Optional fused affine+LeakyReLU input transform.

__device__ __forceinline__ v16h load_a_frag(const float* __restrict__ rowp, int kbase, int koff,
                                            const float* __restrict__ tsc,
                                            const float* __restrict__ tsh) {
  v16h a;
#pragma unroll
  for (int g = 0; g < 2; ++g) {
    int kb = kbase + g * 16 + koff;
    const v4f* p = (const v4f*)(rowp + kb);
    v4f x0 = p[0], x1 = p[1];
    float v[8] = {x0.x, x0.y, x0.z, x0.w, x1.x, x1.y, x1.z, x1.w};
#pragma unroll
    for (int i = 0; i < 8; ++i) {
      float f = v[i];
      if (tsc) f = lrelu(f * tsc[kb + i] + tsh[kb + i]);
      a[g * 8 + i] = (_Float16)f;
    }
  }
  return a;
}

// A-frag for the node update:  val = deg>0 ? nodesum/deg : transform(prev)
__device__ __forceinline__ v16h load_a_frag_sel(const float* __restrict__ nsp,
                                                const float* __restrict__ pvp,
                                                float inv, bool use_ns, int kbase, int koff,
                                                const float* __restrict__ tsc,
                                                const float* __restrict__ tsh) {
  v16h a;
#pragma unroll
  for (int g = 0; g < 2; ++g) {
    int kb = kbase + g * 16 + koff;
    const v4f* pn = (const v4f*)(nsp + kb);
    const v4f* pp = (const v4f*)(pvp + kb);
    v4f n0 = pn[0], n1 = pn[1];
    v4f p0 = pp[0], p1 = pp[1];
    float nv[8] = {n0.x, n0.y, n0.z, n0.w, n1.x, n1.y, n1.z, n1.w};
    float pv[8] = {p0.x, p0.y, p0.z, p0.w, p1.x, p1.y, p1.z, p1.w};
#pragma unroll
    for (int i = 0; i < 8; ++i) {
      float f = pv[i];
      if (tsc) f = lrelu(f * tsc[kb + i] + tsh[kb + i]);
      f = use_ns ? nv[i] * inv : f;
      a[g * 8 + i] = (_Float16)f;
    }
  }
  return a;
}

// 16-bit B-matrix 32x16 layout: lane holds column n = lane%16; lanes 0-15 hold
// K0..7,K16..23, lanes 16-31 hold K8..15,K24..31 (matches A).
__device__ __forceinline__ v16h load_b_frag(const _Float16* __restrict__ Wt, int n, int kbase,
                                            int half) {
  const _Float16* p = Wt + n * 64 + kbase + half * 8;
  v8h lo = *(const v8h*)p, hi = *(const v8h*)(p + 16);
  v16h b;
#pragma unroll
  for (int i = 0; i < 8; ++i) { b[i] = lo[i]; b[8 + i] = hi[i]; }
  return b;
}

// block-level column-stat reduction: LDS ds_add_f32, then one global flush
__device__ __forceinline__ void stat_flush(float* sred, float* colsum, float* colsumsq) {
  __syncthreads();
  if (colsum) {
    int t = threadIdx.x;
    if (t < 64)                unsafeAtomicAdd(&colsum[t], sred[t]);
    else if (t < 128)          unsafeAtomicAdd(&colsumsq[t - 64], sred[t]);
  }
}

// ------------------------------ edge GEMM ----------------------------------
// One wave = one 16-row x 64-col tile.  8 waves / block.
// Y[row,64] = transform(X[gather(row)]) @ W + b   (+ per-column sum/sumsq)

__global__ __launch_bounds__(256) void k_edge_fc(
    const float* __restrict__ X, const int* __restrict__ gather,
    const float* __restrict__ tsc, const float* __restrict__ tsh,
    const _Float16* __restrict__ Wt, const float* __restrict__ bias,
    float* __restrict__ Yout, float* __restrict__ colsum, float* __restrict__ colsumsq,
    int rows) {
  __shared__ float sred[128];
  if (threadIdx.x < 128) sred[threadIdx.x] = 0.f;
  __syncthreads();

  int wave = threadIdx.x >> 5, lane = threadIdx.x & 31;
  int tile = blockIdx.x * 8 + wave;
  int row0 = tile * 16;
  bool active = row0 < rows;
  if (active) {
    int m = lane & 15, half = lane >> 4, koff = half * 8;
    int arow = row0 + m;
    if (gather) arow = gather[arow];
    const float* rowp = X + (long)arow * 64;
    v16h a0 = load_a_frag(rowp, 0, koff, tsc, tsh);
    v16h a1 = load_a_frag(rowp, 32, koff, tsc, tsh);

    int n_lo = lane & 15;
#pragma unroll
    for (int nt = 0; nt < 4; ++nt) {
      int n = nt * 16 + n_lo;
      float bv = bias[n];
      v8f c = {bv, bv, bv, bv, bv, bv, bv, bv};
      v16h b0 = load_b_frag(Wt, n, 0, half);
      v16h b1 = load_b_frag(Wt, n, 32, half);
      c = __builtin_amdgcn_wmma_f32_16x16x32_f16(false, a0, false, b0, (short)0, c, false, false);
      c = __builtin_amdgcn_wmma_f32_16x16x32_f16(false, a1, false, b1, (short)0, c, false, false);

      float s = 0.f, q = 0.f;
#pragma unroll
      for (int r = 0; r < 8; ++r) {
        float v = c[r];
        Yout[(long)(row0 + half * 8 + r) * 64 + n] = v;  // C layout: M=r+half*8, N=n
        s += v;
        q += v * v;
      }
      if (colsum) {
        atomicAdd(&sred[n], s);        // ds_add_f32
        atomicAdd(&sred[64 + n], q);
      }
    }
  }
  stat_flush(sred, colsum, colsumsq);
}

// ------------------------------ node GEMM ----------------------------------
// hbar = deg>0 ? nodesum/deg : transform(prev);  Z = hbar @ Wl + bl
// deg comes from the CSR offsets (off[n+1]-off[n]).

__global__ __launch_bounds__(256) void k_node_fc(
    const float* __restrict__ NS, const int* __restrict__ off,
    const float* __restrict__ Prev, const float* __restrict__ tsc, const float* __restrict__ tsh,
    const _Float16* __restrict__ Wt, const float* __restrict__ bias,
    float* __restrict__ Zout, float* __restrict__ colsum, float* __restrict__ colsumsq,
    int rows) {
  __shared__ float sred[128];
  if (threadIdx.x < 128) sred[threadIdx.x] = 0.f;
  __syncthreads();

  int wave = threadIdx.x >> 5, lane = threadIdx.x & 31;
  int tile = blockIdx.x * 8 + wave;
  int row0 = tile * 16;
  bool active = row0 < rows;
  if (active) {
    int m = lane & 15, half = lane >> 4, koff = half * 8;
    int arow = row0 + m;
    int d = off[arow + 1] - off[arow];
    bool use_ns = d > 0;
    float inv = use_ns ? 1.f / (float)d : 0.f;
    const float* nsp = NS + (long)arow * 64;
    const float* pvp = Prev + (long)arow * 64;
    v16h a0 = load_a_frag_sel(nsp, pvp, inv, use_ns, 0, koff, tsc, tsh);
    v16h a1 = load_a_frag_sel(nsp, pvp, inv, use_ns, 32, koff, tsc, tsh);

    int n_lo = lane & 15;
#pragma unroll
    for (int nt = 0; nt < 4; ++nt) {
      int n = nt * 16 + n_lo;
      float bv = bias[n];
      v8f c = {bv, bv, bv, bv, bv, bv, bv, bv};
      v16h b0 = load_b_frag(Wt, n, 0, half);
      v16h b1 = load_b_frag(Wt, n, 32, half);
      c = __builtin_amdgcn_wmma_f32_16x16x32_f16(false, a0, false, b0, (short)0, c, false, false);
      c = __builtin_amdgcn_wmma_f32_16x16x32_f16(false, a1, false, b1, (short)0, c, false, false);

      float s = 0.f, q = 0.f;
#pragma unroll
      for (int r = 0; r < 8; ++r) {
        float v = c[r];
        Zout[(long)(row0 + half * 8 + r) * 64 + n] = v;
        s += v;
        q += v * v;
      }
      if (colsum) {
        atomicAdd(&sred[n], s);
        atomicAdd(&sred[64 + n], q);
      }
    }
  }
  stat_flush(sred, colsum, colsumsq);
}

// ---------------- bn2 + lrelu + mailbox sum (atomic-free gather) -----------
// One wave per node: lanes cover the 64 columns (2 each); edge rows of Y are
// read fully coalesced (256B per edge); NS row is owned exclusively.

__global__ __launch_bounds__(256) void k_gsum(
    const float* __restrict__ Y, const int* __restrict__ el, const int* __restrict__ off,
    const float* __restrict__ sc, const float* __restrict__ sh, float* __restrict__ NS) {
  int node = blockIdx.x * 8 + (threadIdx.x >> 5);
  if (node >= kN) return;
  int lane = threadIdx.x & 31;
  int c0 = lane * 2, c1 = c0 + 1;
  float s0 = sc[c0], h0 = sh[c0], s1 = sc[c1], h1 = sh[c1];
  int beg = off[node], end = off[node + 1];
  float a0 = 0.f, a1 = 0.f;
  for (int i = beg; i < end; ++i) {
    int e = el[i];
    const float2* p = (const float2*)(Y + (long)e * 64 + c0);
    float2 v = *p;
    a0 += lrelu(v.x * s0 + h0);
    a1 += lrelu(v.y * s1 + h1);
  }
  float2 outv = {a0, a1};
  *(float2*)(NS + (long)node * 64 + c0) = outv;
}

// ------------------------ per-graph max readout ----------------------------
// monotone unsigned key: key(f) = sign(f) ? ~bits : bits|0x80000000

__global__ void k_segmax(const float* __restrict__ Z, const int* __restrict__ gid,
                         unsigned* __restrict__ keys) {
  long i = (long)blockIdx.x * blockDim.x + threadIdx.x;
  if (i < (long)kN * 64) {
    int r = (int)(i >> 6), c = (int)(i & 63);
    unsigned u = __float_as_uint(Z[i]);
    unsigned k = (u & 0x80000000u) ? ~u : (u | 0x80000000u);
    atomicMax(&keys[gid[r] * 64 + c], k);
  }
}

// ------------------------------ MLP head -----------------------------------
// 64x64 -> fc1(128)+bn+lrelu -> fc2(128)+bn+lrelu -> fc_final(10) -> log_softmax
// ~3 MFLOP total: one LDS-resident workgroup.  dynamic LDS: 2 * 64*128 floats.

__global__ __launch_bounds__(128) void k_head(
    const unsigned* __restrict__ keys,
    const float* __restrict__ W1, const float* __restrict__ b1,
    const float* __restrict__ g1, const float* __restrict__ be1,
    const float* __restrict__ W2, const float* __restrict__ b2,
    const float* __restrict__ g2, const float* __restrict__ be2,
    const float* __restrict__ W3, const float* __restrict__ b3,
    float* __restrict__ out) {
  extern __shared__ float smem[];
  float* A = smem;                // 64*128: x (first 4096), later a2
  float* B = smem + 64 * 128;     // 64*128: a1, later logits
  int t = threadIdx.x;

  for (int i = t; i < 64 * 64; i += 128) {       // decode per-graph maxima
    unsigned k = keys[i];
    unsigned u = (k & 0x80000000u) ? (k ^ 0x80000000u) : ~k;
    A[i] = __uint_as_float(u);
  }
  __syncthreads();
  for (int i = t; i < 64 * 128; i += 128) {      // a1 = x @ W1 + b1
    int r = i >> 7, j = i & 127;
    float acc = b1[j];
    for (int k2 = 0; k2 < 64; ++k2) acc += A[r * 64 + k2] * W1[k2 * 128 + j];
    B[i] = acc;
  }
  __syncthreads();
  if (t < 128) {                                 // bn_fc1 + lrelu (batch = 64 rows)
    float s = 0.f, q = 0.f;
    for (int r = 0; r < 64; ++r) { float v = B[r * 128 + t]; s += v; q += v * v; }
    float mean = s / 64.f, var = q / 64.f - mean * mean;
    float sc = g1[t] * rsqrtf(var + kEps), sf = be1[t] - mean * sc;
    for (int r = 0; r < 64; ++r) B[r * 128 + t] = lrelu(B[r * 128 + t] * sc + sf);
  }
  __syncthreads();
  for (int i = t; i < 64 * 128; i += 128) {      // a2 = a1 @ W2 + b2
    int r = i >> 7, j = i & 127;
    float acc = b2[j];
    for (int k2 = 0; k2 < 128; ++k2) acc += B[r * 128 + k2] * W2[k2 * 128 + j];
    A[i] = acc;
  }
  __syncthreads();
  if (t < 128) {                                 // bn_fc2 + lrelu
    float s = 0.f, q = 0.f;
    for (int r = 0; r < 64; ++r) { float v = A[r * 128 + t]; s += v; q += v * v; }
    float mean = s / 64.f, var = q / 64.f - mean * mean;
    float sc = g2[t] * rsqrtf(var + kEps), sf = be2[t] - mean * sc;
    for (int r = 0; r < 64; ++r) A[r * 128 + t] = lrelu(A[r * 128 + t] * sc + sf);
  }
  __syncthreads();
  for (int i = t; i < 64 * kOut; i += 128) {     // logits
    int r = i / kOut, o = i % kOut;
    float acc = b3[o];
    for (int k2 = 0; k2 < 128; ++k2) acc += A[r * 128 + k2] * W3[k2 * kOut + o];
    B[i] = acc;
  }
  __syncthreads();
  if (t < 64) {                                  // log_softmax per row
    float mx = -INFINITY;
    for (int o = 0; o < kOut; ++o) mx = fmaxf(mx, B[t * kOut + o]);
    float se = 0.f;
    for (int o = 0; o < kOut; ++o) se += expf(B[t * kOut + o] - mx);
    float lse = mx + logf(se);
    for (int o = 0; o < kOut; ++o) out[t * kOut + o] = B[t * kOut + o] - lse;
  }
}

// ------------------------------- launcher ----------------------------------

extern "C" void kernel_launch(void* const* d_in, const int* in_sizes, int n_in,
                              void* d_out, int out_size, void* d_ws, size_t ws_size,
                              hipStream_t stream) {
  (void)in_sizes; (void)n_in; (void)out_size; (void)ws_size;
  const float* node_feats = (const float*)d_in[0];
  const int*   src        = (const int*)d_in[1];
  const int*   dst        = (const int*)d_in[2];
  const int*   gid        = (const int*)d_in[3];
  // d_in[4] = num_graphs (constant 64)
  auto P = [&](int i) { return (const float*)d_in[i]; };

  // ---- workspace layout (≈ 218.5 MB) ----
  char* w = (char*)d_ws;
  float*     Y     = (float*)w;      w += (size_t)kE * 64 * 4;   // edge activations
  float*     Z     = (float*)w;      w += (size_t)kN * 64 * 4;   // node layer output
  float*     NS    = (float*)w;      w += (size_t)kN * 64 * 4;   // mailbox sums
  int*       CNT   = (int*)w;        w += (size_t)kN * 4;
  int*       OFF   = (int*)w;        w += (size_t)(kN + 1) * 4;
  int*       CUR   = (int*)w;        w += (size_t)(kN + 1) * 4;
  int*       EL    = (int*)w;        w += (size_t)kE * 4;        // edge ids by dst
  unsigned*  KEYS  = (unsigned*)w;   w += (size_t)kG * 64 * 4;
  float*     STATS = (float*)w;      w += (size_t)9 * 256 * 4;   // 9 bn slots
  _Float16*  WT    = (_Float16*)w;                               // 9 x 64x64 f16
  auto slot = [&](int s) { return STATS + s * 256; };  // [cs|cq|scale|shift] x64

  const int T = 256;
  auto zgrid = [&](int n) { return (n + T - 1) / T; };

  // ---- prep: stats/keys/hist zero, weight f16-transpose, CSR build ----
  k_zero<<<zgrid(9 * 256), T, 0, stream>>>(STATS, 9 * 256);
  k_zero<<<zgrid(kG * 64), T, 0, stream>>>((float*)KEYS, kG * 64);
  k_zero<<<zgrid(kN), T, 0, stream>>>((float*)CNT, kN);
  for (int l = 0; l < 3; ++l) {
    int mb = (l == 0) ? 5 : (l == 1) ? 17 : 29;  // mp{1,2,3} base index
    k_w_to_f16t<<<16, T, 0, stream>>>(P(mb + 0), WT + (l * 3 + 0) * 4096);
    k_w_to_f16t<<<16, T, 0, stream>>>(P(mb + 4), WT + (l * 3 + 1) * 4096);
    k_w_to_f16t<<<16, T, 0, stream>>>(P(mb + 8), WT + (l * 3 + 2) * 4096);
  }
  k_hist<<<zgrid(kE), T, 0, stream>>>(dst, CNT);
  k_scan<<<1, 1024, 0, stream>>>(CNT, OFF, CUR);
  k_fill<<<zgrid(kE), T, 0, stream>>>(dst, CUR, EL);

  const int edgeBlocks = kE / 16 / 8;            // 5000 (exact)
  const int nodeBlocks = (kN / 16 + 7) / 8;      // 782
  const int gsumBlocks = (kN + 7) / 8;           // 12500 (1 wave per node)
  const int smaxBlocks = (int)(((long)kN * 64 + 255) / 256);

  // ---- three message-passing layers ----
  for (int l = 0; l < 3; ++l) {
    int mb = (l == 0) ? 5 : (l == 1) ? 17 : 29;  // W1 b1 g1 be1 W2 b2 g2 be2 Wl bl
    float* st1 = slot(l * 3 + 0);
    float* st2 = slot(l * 3 + 1);
    float* stp = slot(l * 3 + 2);
    const float* prev; const float* ptsc; const float* ptsh;
    if (l == 0) { prev = node_feats; ptsc = nullptr; ptsh = nullptr; }
    else { prev = Z; float* ps = slot((l - 1) * 3 + 2); ptsc = ps + 128; ptsh = ps + 192; }
    const _Float16* W1t = WT + (l * 3 + 0) * 4096;
    const _Float16* W2t = WT + (l * 3 + 1) * 4096;
    const _Float16* Wlt = WT + (l * 3 + 2) * 4096;

    // fc1 on gathered edges (fused prev-layer bn+lrelu on load), Y = raw pre-bn1
    k_edge_fc<<<edgeBlocks, T, 0, stream>>>(prev, src, ptsc, ptsh, W1t, P(mb + 1),
                                            Y, st1, st1 + 64, kE);
    k_bn_finalize<<<1, 64, 0, stream>>>(st1, st1 + 64, P(mb + 2), P(mb + 3),
                                        st1 + 128, st1 + 192, (float)kE);
    // fc2 (fused bn1+lrelu on load), in-place Y = raw pre-bn2
    k_edge_fc<<<edgeBlocks, T, 0, stream>>>(Y, nullptr, st1 + 128, st1 + 192, W2t, P(mb + 5),
                                            Y, st2, st2 + 64, kE);
    k_bn_finalize<<<1, 64, 0, stream>>>(st2, st2 + 64, P(mb + 6), P(mb + 7),
                                        st2 + 128, st2 + 192, (float)kE);
    // bn2 + lrelu + mailbox sum via CSR gather (no atomics)
    k_gsum<<<gsumBlocks, T, 0, stream>>>(Y, EL, OFF, st2 + 128, st2 + 192, NS);
    // node update GEMM (+ post-bn stats for layers 0,1)
    bool needPost = (l < 2);
    k_node_fc<<<nodeBlocks, T, 0, stream>>>(NS, OFF, prev, ptsc, ptsh, Wlt, P(mb + 9), Z,
                                            needPost ? stp : nullptr,
                                            needPost ? stp + 64 : nullptr, kN);
    if (needPost) {
      int pb = (l == 0) ? 15 : 27;               // post{1,2}: gamma, beta
      k_bn_finalize<<<1, 64, 0, stream>>>(stp, stp + 64, P(pb), P(pb + 1),
                                          stp + 128, stp + 192, (float)kN);
    }
  }

  // ---- readout + head ----
  k_segmax<<<smaxBlocks, T, 0, stream>>>(Z, gid, KEYS);
  k_head<<<1, 128, 2 * 64 * 128 * sizeof(float), stream>>>(
      KEYS, P(39), P(40), P(41), P(42), P(43), P(44), P(45), P(46), P(47), P(48),
      (float*)d_out);
}
